// PointNetSetAbstraction_12008728560018
// MI455X (gfx1250) — matrix operand order repaired and checked by reading, not compiled, not run on hardware
//
#include <hip/hip_runtime.h>
#include <hip/hip_bf16.h>

// ---------------- problem constants ----------------
#define B_    4
#define N_    16384
#define S_    1024
#define K_    32
#define DIN_  16
#define MTOT  (B_ * S_ * K_)   // 131072 rows
#define R2_   0.25f            // RADIUS^2
#define EPS_  1e-5f

typedef __attribute__((ext_vector_type(16))) _Float16 v16h;
typedef __attribute__((ext_vector_type(8)))  _Float16 v8h;
typedef __attribute__((ext_vector_type(8)))  float    v8f;

// ---------------------------------------------------------------------------
// K0: gather new_xyz (B,S,3) into ws and write new_xyz output (B,3,S)
// ---------------------------------------------------------------------------
__global__ __launch_bounds__(256) void gather_newxyz(
    const float* __restrict__ xyz, const int* __restrict__ fps,
    float* __restrict__ nx, float* __restrict__ out)
{
  int i = blockIdx.x * 256 + threadIdx.x;           // [0, B*S)
  if (i >= B_ * S_) return;
  int b = i / S_, s = i % S_;
  int id = fps[i];
  const float* xb = xyz + (size_t)b * 3 * N_;
#pragma unroll
  for (int c = 0; c < 3; ++c) {
    float v = xb[(size_t)c * N_ + id];
    nx[(size_t)i * 3 + c] = v;
    out[(size_t)b * 3 * S_ + (size_t)c * S_ + s] = v;
  }
}

// ---------------------------------------------------------------------------
// K1: pack all three weight matrices to f16, CIN padded (19 -> 32)
// layout in wh (halfs): W0 [64][32] at 0, W1 [64][64] at 2048, W2 [128][64] at 6144
// ---------------------------------------------------------------------------
__global__ __launch_bounds__(256) void pack_weights(
    const float* __restrict__ W0, const float* __restrict__ W1,
    const float* __restrict__ W2, _Float16* __restrict__ wh)
{
  int t = blockIdx.x * 256 + threadIdx.x;           // 8192 threads
  if (t < 64 * 32) {
    int o = t >> 5, c = t & 31;
    wh[t] = (_Float16)((c < 19) ? W0[o * 19 + c] : 0.f);
  }
  if (t < 64 * 64)  wh[2048 + t] = (_Float16)W1[t];
  if (t < 128 * 64) wh[6144 + t] = (_Float16)W2[t];
}

// ---------------------------------------------------------------------------
// K2: ball query — one wave32 per query point; ballot+prefix picks the first
// K in-radius indices in ascending order (== reference top_k(-cand) trick).
// ---------------------------------------------------------------------------
__global__ __launch_bounds__(256) void ball_query(
    const float* __restrict__ xyz, const float* __restrict__ nx,
    int* __restrict__ idx)
{
  int wave = threadIdx.x >> 5;
  int lane = threadIdx.x & 31;
  int q = blockIdx.x * 8 + wave;                    // [0, B*S)
  int b = q / S_;
  float qx = nx[(size_t)q * 3 + 0];
  float qy = nx[(size_t)q * 3 + 1];
  float qz = nx[(size_t)q * 3 + 2];
  const float* xb = xyz + (size_t)b * 3 * N_;
  int* out = idx + (size_t)q * K_;

  int cnt = 0;
  for (int base = 0; base < N_ && cnt < K_; base += 32) {
    int n = base + lane;
    float dx = xb[n] - qx;
    float dy = xb[N_ + n] - qy;
    float dz = xb[2 * N_ + n] - qz;
    bool hit = (dx * dx + dy * dy + dz * dz) <= R2_;
    unsigned mask = (unsigned)__ballot(hit);
    int pos = cnt + __popc(mask & ((1u << lane) - 1u));
    if (hit && pos < K_) out[pos] = n;
    cnt += __popc(mask);
  }
  if (cnt > K_) cnt = K_;
  int firstv = 0;
  if (cnt > 0) firstv = out[0];                     // same-wave st->ld, in order
  if (lane >= cnt) out[lane] = firstv;              // pad (0 if no hits)
}

// ---------------------------------------------------------------------------
// K3: group + concat [g_xyz | g_feat | zero-pad] -> f16 A0 buffer, M x 32
// ---------------------------------------------------------------------------
__global__ __launch_bounds__(256) void group_pack(
    const float* __restrict__ xyz, const float* __restrict__ pts,
    const int* __restrict__ idx, const float* __restrict__ nx,
    _Float16* __restrict__ A)
{
  int i = blockIdx.x * 256 + threadIdx.x;           // [0, MTOT)
  int bs = i / K_;
  int b = bs / S_;
  int id = idx[i];
  const float* xb = xyz + (size_t)b * 3 * N_;
  const float* pb = pts + (size_t)b * DIN_ * N_;
  _Float16* row = A + (size_t)i * 32;
  row[0] = (_Float16)(xb[id]           - nx[(size_t)bs * 3 + 0]);
  row[1] = (_Float16)(xb[N_ + id]      - nx[(size_t)bs * 3 + 1]);
  row[2] = (_Float16)(xb[2 * N_ + id]  - nx[(size_t)bs * 3 + 2]);
#pragma unroll
  for (int c = 0; c < DIN_; ++c) row[3 + c] = (_Float16)pb[(size_t)c * N_ + id];
#pragma unroll
  for (int c = 19; c < 32; ++c) row[c] = (_Float16)0.f;
}

// ---------------------------------------------------------------------------
// GEMM: Y[M,COUT] = A[M,CIN](f16) x W[COUT,CIN]^T(f16), f32 accumulate via
// v_wmma_f32_16x16x32_f16. One wave owns a 16 x COUT strip. Fused epilogue
// produces deterministic per-block per-channel (sum, sumsq) partials.
// ---------------------------------------------------------------------------
template<int CIN, int COUT>
__global__ __launch_bounds__(256) void gemm_wmma(
    const _Float16* __restrict__ A, const _Float16* __restrict__ W,
    float* __restrict__ Y, float* __restrict__ part)
{
  constexpr int NT = COUT / 16;
  constexpr int KB = CIN / 32;
  __shared__ float s_sum[8][COUT];
  __shared__ float s_sq[8][COUT];

  const int tid    = threadIdx.x;
  const int wave   = tid >> 5;
  const int lane   = tid & 31;
  const int lane16 = lane & 15;
  const int hi     = lane >> 4;                     // 0 | 1 half-wave
  const int m0     = (blockIdx.x * 8 + wave) << 4;
  const bool active = (m0 < MTOT);                  // wave-uniform

  v8f acc[NT] = {};

  if (active) {
#pragma unroll
    for (int kb = 0; kb < KB; ++kb) {
      // A fragment: lane holds row m0+lane16, halfs {kb*32+ka..+7, +16..+23}
      const _Float16* ap = A + (size_t)(m0 + lane16) * CIN + kb * 32 + (hi ? 8 : 0);
      v8h alo = *(const v8h*)ap;
      v8h ahi = *(const v8h*)(ap + 16);
      v16h af;
#pragma unroll
      for (int i = 0; i < 8; ++i) { af[i] = alo[i]; af[i + 8] = ahi[i]; }

#pragma unroll
      for (int nt = 0; nt < NT; ++nt) {
        // B fragment: lane holds column o = nt*16+lane16, halfs {kb*32+ka..+7, +16..+23}
        const _Float16* bp = W + (size_t)(nt * 16 + lane16) * CIN + kb * 32 + (hi ? 8 : 0);
        v8h blo = *(const v8h*)bp;
        v8h bhi = *(const v8h*)(bp + 16);
        v16h bf;
#pragma unroll
        for (int i = 0; i < 8; ++i) { bf[i] = blo[i]; bf[i + 8] = bhi[i]; }
        acc[nt] = __builtin_amdgcn_wmma_f32_16x16x32_f16(
            false, af, false, bf, (short)0, acc[nt], false, false);
      }
    }
  }

  // Epilogue: store f32 result + deterministic per-channel stats partials
#pragma unroll
  for (int nt = 0; nt < NT; ++nt) {
    float s = 0.f, q = 0.f;
    if (active) {
#pragma unroll
      for (int v = 0; v < 8; ++v) {
        float f = acc[nt][v];
        int row = m0 + v + (hi << 3);               // D layout: M = v + 8*hi
        Y[(size_t)row * COUT + nt * 16 + lane16] = f;
        s += f; q += f * f;
      }
    }
    s += __shfl_xor(s, 16, 32);                     // merge the two half-waves
    q += __shfl_xor(q, 16, 32);
    if (hi == 0) { s_sum[wave][nt * 16 + lane16] = s; s_sq[wave][nt * 16 + lane16] = q; }
  }
  __syncthreads();
  if (tid < COUT) {
    float s = 0.f, q = 0.f;
#pragma unroll
    for (int w = 0; w < 8; ++w) { s += s_sum[w][tid]; q += s_sq[w][tid]; }
    part[(size_t)blockIdx.x * (2 * COUT) + tid]        = s;
    part[(size_t)blockIdx.x * (2 * COUT) + COUT + tid] = q;
  }
}

// ---------------------------------------------------------------------------
// Reduce per-block partials -> stats[2*COUT] (fixed order, deterministic)
// ---------------------------------------------------------------------------
__global__ __launch_bounds__(256) void reduce_stats(
    const float* __restrict__ part, float* __restrict__ stats,
    int nblocks, int cout2)
{
  int t = threadIdx.x;
  if (t < cout2) {
    float a = 0.f;
    for (int i = 0; i < nblocks; ++i) a += part[(size_t)i * cout2 + t];
    stats[t] = a;
  }
}

// ---------------------------------------------------------------------------
// BN (global mean/var) + ReLU + repack to f16 for next layer's A operand
// ---------------------------------------------------------------------------
__global__ __launch_bounds__(256) void bnrelu_pack(
    const float* __restrict__ Y, const float* __restrict__ stats,
    const float* __restrict__ gamma, const float* __restrict__ beta,
    _Float16* __restrict__ Aout, int cout)
{
  size_t i = (size_t)blockIdx.x * 256 + threadIdx.x;  // [0, MTOT*cout)
  int c = (int)(i % cout);
  float invn = 1.0f / (float)MTOT;
  float mu  = stats[c] * invn;
  float var = stats[cout + c] * invn - mu * mu;
  float r   = rsqrtf(var + EPS_);
  float v   = (Y[i] - mu) * r * gamma[c] + beta[c];
  Aout[i] = (_Float16)fmaxf(v, 0.f);
}

// ---------------------------------------------------------------------------
// Final: BN + ReLU + max over K neighbors; write new_points as (B,128,S)
// ---------------------------------------------------------------------------
__global__ __launch_bounds__(256) void bn_relu_maxk(
    const float* __restrict__ Y, const float* __restrict__ stats,
    const float* __restrict__ gamma, const float* __restrict__ beta,
    float* __restrict__ out)
{
  int i = blockIdx.x * 256 + threadIdx.x;           // [0, B*S*128)
  int o  = i & 127;
  int bs = i >> 7;
  int b  = bs / S_;
  int s  = bs % S_;
  float invn = 1.0f / (float)MTOT;
  float mu  = stats[o] * invn;
  float var = stats[128 + o] * invn - mu * mu;
  float r   = rsqrtf(var + EPS_);
  float g = gamma[o], be = beta[o];
  float m = 0.f;                                    // relu outputs are >= 0
  const float* yb = Y + (size_t)bs * K_ * 128 + o;
  for (int k = 0; k < K_; ++k) {
    float y = yb[(size_t)k * 128];
    y = fmaxf((y - mu) * r * g + be, 0.f);
    m = fmaxf(m, y);
  }
  out[(size_t)B_ * 3 * S_ + (size_t)b * 128 * S_ + (size_t)o * S_ + s] = m;
}

// ---------------------------------------------------------------------------
extern "C" void kernel_launch(void* const* d_in, const int* in_sizes, int n_in,
                              void* d_out, int out_size, void* d_ws, size_t ws_size,
                              hipStream_t stream)
{
  (void)in_sizes; (void)n_in; (void)out_size; (void)ws_size;
  const float* xyz = (const float*)d_in[0];
  const float* pts = (const float*)d_in[1];
  const int*   fps = (const int*)  d_in[2];
  const float* W0  = (const float*)d_in[3];
  const float* g0  = (const float*)d_in[4];
  const float* b0  = (const float*)d_in[5];
  const float* W1  = (const float*)d_in[6];
  const float* g1  = (const float*)d_in[7];
  const float* b1  = (const float*)d_in[8];
  const float* W2  = (const float*)d_in[9];
  const float* g2  = (const float*)d_in[10];
  const float* b2  = (const float*)d_in[11];
  float* out = (float*)d_out;

  // workspace layout (bytes, all 256B aligned)
  char* ws = (char*)d_ws;
  float*    nx    = (float*)   (ws + 0);          //  49152: new_xyz (B*S*3) f32
  int*      idx   = (int*)     (ws + 49152);      // 524288: ball-query idx
  _Float16* wh    = (_Float16*)(ws + 573440);     //  28672: packed f16 weights
  float*    stats = (float*)   (ws + 602112);     //   2048: 3 layers' (sum,sq)
  float*    part  = (float*)   (ws + 604160);     //   1 MB: per-block partials
  _Float16* Ab    = (_Float16*)(ws + 1652736);    //  16 MB: activation f16 (M x <=64)
  float*    Yb    = (float*)   (ws + 18429952);   //  64 MB: GEMM output f32 (M x <=128)

  gather_newxyz<<<16, 256, 0, stream>>>(xyz, fps, nx, out);
  pack_weights <<<32, 256, 0, stream>>>(W0, W1, W2, wh);
  ball_query   <<<B_ * S_ / 8, 256, 0, stream>>>(xyz, nx, idx);
  group_pack   <<<MTOT / 256, 256, 0, stream>>>(xyz, pts, idx, nx, Ab);

  const int GB = MTOT / 16 / 8;                   // 1024 GEMM blocks

  // layer 0: 19(pad 32) -> 64
  gemm_wmma<32, 64><<<GB, 256, 0, stream>>>(Ab, wh, Yb, part);
  reduce_stats<<<1, 256, 0, stream>>>(part, stats + 0, GB, 128);
  bnrelu_pack<<<(MTOT * 64) / 256, 256, 0, stream>>>(Yb, stats + 0, g0, b0, Ab, 64);

  // layer 1: 64 -> 64
  gemm_wmma<64, 64><<<GB, 256, 0, stream>>>(Ab, wh + 2048, Yb, part);
  reduce_stats<<<1, 256, 0, stream>>>(part, stats + 128, GB, 128);
  bnrelu_pack<<<(MTOT * 64) / 256, 256, 0, stream>>>(Yb, stats + 128, g1, b1, Ab, 64);

  // layer 2: 64 -> 128
  gemm_wmma<64, 128><<<GB, 256, 0, stream>>>(Ab, wh + 6144, Yb, part);
  reduce_stats<<<1, 256, 0, stream>>>(part, stats + 256, GB, 256);
  bn_relu_maxk<<<(B_ * S_ * 128) / 256, 256, 0, stream>>>(Yb, stats + 256, g2, b2, out);
}
